// TopKSparseAttention_70300024701602
// MI455X (gfx1250) — matrix operand order repaired and checked by reading, not compiled, not run on hardware
//
#include <hip/hip_runtime.h>
#include <hip/hip_bf16.h>

// ---------------- problem constants ----------------
#define B_    2
#define T_    2048
#define E_    1024
#define H_    16
#define D_    64
#define E3_   3072
#define KTOP_ 1433           // max(1, int(0.7 * 2048))
#define SROW  2052           // 2048 + 4 pad floats -> LDS bank-conflict avoidance
#define SB    40             // padded LDS row stride (elems) for staged W tiles

typedef __attribute__((ext_vector_type(16))) __bf16 v16bf;
typedef __attribute__((ext_vector_type(8)))  __bf16 v8bf;
typedef __attribute__((ext_vector_type(8)))  float  v8f;
typedef int v4i_ __attribute__((vector_size(16)));   // matches builtin's int4 pointee

// ---------------- CDNA5 async global->LDS copy (guarded probe) ----------------
#if defined(__has_builtin)
#  if __has_builtin(__builtin_amdgcn_global_load_async_to_lds_b128)
#    define ASYNC_COPY 1
#  endif
#endif
#ifndef ASYNC_COPY
#  define ASYNC_COPY 0
#endif

#if ASYNC_COPY
#  if __has_builtin(__builtin_amdgcn_s_wait_asynccnt)
#    define WAIT_ASYNC(nn) __builtin_amdgcn_s_wait_asynccnt(nn)
#  else
#    define WAIT_ASYNC(nn) asm volatile("s_wait_asynccnt %0" ::"n"(nn) : "memory")
#  endif
#else
#  define WAIT_ASYNC(nn) do {} while (0)
#endif

// copy one 16-byte segment global -> LDS
static __device__ __forceinline__ void stage16(const __bf16* g, __bf16* l) {
#if ASYNC_COPY
  __builtin_amdgcn_global_load_async_to_lds_b128(
      (__attribute__((address_space(1))) v4i_*)(unsigned long long)g,
      (__attribute__((address_space(3))) v4i_*)(unsigned)(unsigned long long)l,
      0, 0);
#else
  *(v8bf*)l = *(const v8bf*)g;
#endif
}

// ---------------- WMMA fragment helpers (per CDNA5 ISA 7.12.2 tables) ----------------
// 16-bit A 16x32: lanes 0-15 hold M=lane, K {0..7,16..23}; lanes 16-31 hold K {8..15,24..31}
static __device__ __forceinline__ v16bf ld_a16(const __bf16* A, long lda, int lane) {
  const int m = lane & 15, hi = (lane >> 4) & 1;
  const __bf16* p = A + (long)m * lda + hi * 8;
  v8bf a0 = *(const v8bf*)(p);
  v8bf a1 = *(const v8bf*)(p + 16);
  return __builtin_shufflevector(a0, a1, 0,1,2,3,4,5,6,7,8,9,10,11,12,13,14,15);
}

// 16-bit B 32x16 from a row-major [N,K] matrix (B[k][n] = W[n][k]):
// lanes 0-15: n=lane, K {0..7,16..23}; lanes 16-31: n=lane-16, K {8..15,24..31}
static __device__ __forceinline__ v16bf ld_b16(const __bf16* Wnk, long ldw, int lane) {
  const int n = lane & 15, hi = (lane >> 4) & 1;
  const __bf16* p = Wnk + (long)n * ldw + hi * 8;
  v8bf a0 = *(const v8bf*)(p);
  v8bf a1 = *(const v8bf*)(p + 16);
  return __builtin_shufflevector(a0, a1, 0,1,2,3,4,5,6,7,8,9,10,11,12,13,14,15);
}

// same B layout but from LDS with 16-byte loads (row stride need only be 16B aligned)
static __device__ __forceinline__ v16bf ld_b16_lds(const __bf16* Wnk, int ldr, int lane) {
  const int n = lane & 15, hi = (lane >> 4) & 1;
  const __bf16* p = Wnk + n * ldr + hi * 8;
  v8bf a0 = *(const v8bf*)(p);
  v8bf a1 = *(const v8bf*)(p + 16);
  return __builtin_shufflevector(a0, a1, 0,1,2,3,4,5,6,7,8,9,10,11,12,13,14,15);
}

// A fragment sourced from fp32 LDS row-major tile (softmax probabilities)
static __device__ __forceinline__ v16bf ld_a_lds_f32(const float* base, int ldr, int lane) {
  const int m = lane & 15, hi = (lane >> 4) & 1;
  const float* p = base + m * ldr + hi * 8;
  v16bf a;
#pragma unroll
  for (int j = 0; j < 8; ++j) { a[j] = (__bf16)p[j]; a[j + 8] = (__bf16)p[16 + j]; }
  return a;
}

static __device__ __forceinline__ v8f wmma_bf16(v16bf a, v16bf b, v8f c) {
  return __builtin_amdgcn_wmma_f32_16x16x32_bf16(false, a, false, b, (short)0, c, false, false);
}

// monotonic float -> uint key (ascending)
static __device__ __forceinline__ unsigned f2key(float f) {
  unsigned u = __float_as_uint(f);
  return (u & 0x80000000u) ? ~u : (u | 0x80000000u);
}

// ---------------- fp32 -> bf16 conversion ----------------
__global__ void cvt_kernel(const float* __restrict__ s, __bf16* __restrict__ d, size_t n) {
  size_t i = (size_t)blockIdx.x * blockDim.x + threadIdx.x;
  const size_t stride = (size_t)gridDim.x * blockDim.x;
  for (; i < n; i += stride) d[i] = (__bf16)s[i];
}

// ---------------- V transpose: qkv[b,t,2E + h*64 + d] -> vt[b,h,d,t] ----------------
__global__ void vT_kernel(const __bf16* __restrict__ qkv, __bf16* __restrict__ vt) {
  size_t i = (size_t)blockIdx.x * blockDim.x + threadIdx.x;   // over B*H*D*T, t fastest
  const int t = (int)(i & (T_ - 1));
  size_t r = i >> 11;
  const int d = (int)(r & (D_ - 1)); r >>= 6;
  const int h = (int)(r & (H_ - 1));
  const int b = (int)(r >> 4);
  vt[i] = qkv[((size_t)(b * T_ + t)) * E3_ + 2 * E_ + h * D_ + d];
}

// ---------------- WMMA GEMM: C[M,N] = A[M,K] * W[N,K]^T + bias ----------------
// 256 threads = 8 waves; block tile 128(M) x 64(N); wave w owns rows [w*16, w*16+16).
// The 64x32 W chunk is staged once per block into LDS (double-buffered, async
// global->LDS when available) and reused by all 8 waves.
template <int F32OUT>
__global__ void __launch_bounds__(256)
gemm_kernel(const __bf16* __restrict__ A, long lda,
            const __bf16* __restrict__ W, long ldw,
            const float* __restrict__ bias,
            void* __restrict__ Cout, long ldc, int K) {
  __shared__ __bf16 wtile[2][64 * SB];    // 2 x 64 rows x 32 elems (padded to SB)

  const int lane = threadIdx.x & 31;
  const int wave = threadIdx.x >> 5;
  const long n0 = (long)blockIdx.x * 64;
  const long m0 = (long)blockIdx.y * 128 + wave * 16;
  const __bf16* Ab = A + m0 * lda;

  // staging coords: 256 threads x one 16B segment = the whole 64x32 chunk
  const int srow = threadIdx.x >> 2;
  const int scol = (threadIdx.x & 3) * 8;
  const __bf16* wsrc = W + (n0 + srow) * ldw + scol;
  __bf16* wdst0 = &wtile[0][srow * SB + scol];
  __bf16* wdst1 = &wtile[1][srow * SB + scol];

  stage16(wsrc, wdst0);                    // prologue: chunk 0 -> buffer 0

  v8f acc[4] = {};
  const int KC = K / 32;
  for (int kc = 0; kc < KC; ++kc) {
    const int cur = kc & 1;
    if (kc + 1 < KC) {                     // prefetch next chunk into other buffer
      stage16(wsrc + (kc + 1) * 32, cur ? wdst0 : wdst1);
      WAIT_ASYNC(1);                       // current chunk complete, next in flight
    } else {
      WAIT_ASYNC(0);
    }
    __syncthreads();

    const int k0 = kc * 32;
    __builtin_prefetch((const void*)(Ab + (long)(lane & 15) * lda + k0 + 128), 0, 0);
    v16bf a = ld_a16(Ab + k0, lda, lane);
    const __bf16* wb = &wtile[cur][0];
#pragma unroll
    for (int nn = 0; nn < 4; ++nn)
      acc[nn] = wmma_bf16(a, ld_b16_lds(wb + nn * 16 * SB, SB, lane), acc[nn]);
    __syncthreads();                       // all waves done reading before overwrite
  }

  // epilogue: C/D layout -> VGPR r: lanes 0-15 M=r, lanes 16-31 M=r+8; N=lane&15
  const int n = lane & 15, hi = (lane >> 4) & 1;
#pragma unroll
  for (int nn = 0; nn < 4; ++nn) {
    const long col = n0 + nn * 16 + n;
    const float bv = bias[col];
#pragma unroll
    for (int r = 0; r < 8; ++r) {
      const long row = m0 + hi * 8 + r;
      const float v = acc[nn][r] + bv;
      if (F32OUT) ((float*)Cout)[row * ldc + col] = v;
      else        ((__bf16*)Cout)[row * ldc + col] = (__bf16)v;
    }
  }
}

// ---------------- fused top-k attention ----------------
// One block (4 waves) per (b, h, 16-row q tile). Whole score row tile lives in LDS
// (16 x 2052 fp32 = 131 KB; CDNA5 WGP has 320 KB LDS).
// exp(-10000 - rowmax) underflows to 0 in fp32, so the reference's NEG_FILL+softmax
// is exactly "softmax restricted to scores >= k-th largest"; that threshold is found
// exactly with a 4x8-bit radix select on the monotonic uint key.
__global__ void __launch_bounds__(128)
attn_kernel(const __bf16* __restrict__ qkv, const __bf16* __restrict__ vt,
            __bf16* __restrict__ ctx) {
  extern __shared__ char smem[];
  float*    sc     = (float*)smem;                                   // [16][SROW]
  float*    red    = (float*)(smem + (size_t)16 * SROW * 4);         // [4][16][64]
  int*      hist   = (int*)(smem + (size_t)16 * SROW * 4 + 16384);   // [4][256]
  float*    rowinv = (float*)((char*)hist + 4096);                   // [16]
  unsigned* selP   = (unsigned*)(rowinv + 16);                       // [4]
  int*      selR   = (int*)(selP + 4);                               // [4]

  const int lane = threadIdx.x & 31;
  const int wave = threadIdx.x >> 5;
  const int t0 = blockIdx.x * 16;
  const int h  = blockIdx.y;
  const int b  = blockIdx.z;
  const int n  = lane & 15, hi = (lane >> 4) & 1;

  const __bf16* Aq = qkv + ((long)(b * T_ + t0)) * E3_ + h * D_;        // Q tile base
  const __bf16* Kb = qkv + ((long)b * T_) * E3_ + E_ + h * D_;          // K rows base
  const __bf16* Vb = vt + ((long)(b * H_ + h) * D_) * T_;               // V^T [D][T]

  // ---- phase 1: S = (Q K^T) / sqrt(D), streamed into LDS ----
  const v16bf aq0 = ld_a16(Aq,      E3_, lane);
  const v16bf aq1 = ld_a16(Aq + 32, E3_, lane);
  for (int nt = wave; nt < T_ / 16; nt += 4) {
    const __bf16* Kp = Kb + (long)nt * 16 * E3_;
    v8f acc = {};
    acc = wmma_bf16(aq0, ld_b16(Kp,      E3_, lane), acc);
    acc = wmma_bf16(aq1, ld_b16(Kp + 32, E3_, lane), acc);
#pragma unroll
    for (int r = 0; r < 8; ++r)
      sc[(hi * 8 + r) * SROW + nt * 16 + n] = acc[r] * 0.125f;  // D^-0.5 = 1/8
  }
  __syncthreads();

  // ---- phase 2: per-row exact k-th-largest threshold + masked softmax ----
  int* myhist = hist + wave * 256;
  for (int rr = 0; rr < 4; ++rr) {
    const int row = wave * 4 + rr;
    float* scrow = sc + row * SROW;

    unsigned prefix = 0, maskhi = 0;
    int remaining = KTOP_;
    for (int pass = 0; pass < 4; ++pass) {
      const int shift = 24 - 8 * pass;
      for (int i = lane; i < 256; i += 32) myhist[i] = 0;
      __syncthreads();
      for (int c = lane; c < T_; c += 32) {
        const unsigned key = f2key(scrow[c]);
        if ((key & maskhi) == prefix) atomicAdd(&myhist[(key >> shift) & 255], 1);
      }
      __syncthreads();
      if (lane == 0) {
        int cum = 0, bin = 0;
        for (int bb = 255; bb >= 0; --bb) {
          const int cnt = myhist[bb];
          if (cum + cnt >= remaining) { bin = bb; break; }
          cum += cnt;
        }
        selP[wave] = prefix | ((unsigned)bin << shift);
        selR[wave] = remaining - cum;
      }
      __syncthreads();
      prefix = selP[wave];
      remaining = selR[wave];
      maskhi |= 0xFFu << shift;
    }
    const unsigned thr = prefix;   // exact key of the KTOP-th largest score

    // row max (wave butterfly)
    float m = -3.0e38f;
    for (int c = lane; c < T_; c += 32) m = fmaxf(m, scrow[c]);
#pragma unroll
    for (int off = 16; off >= 1; off >>= 1) m = fmaxf(m, __shfl_xor(m, off, 32));

    // masked exp in place; defer 1/sum normalization to the ctx epilogue
    float lsum = 0.f;
    for (int c = lane; c < T_; c += 32) {
      const float s = scrow[c];
      const float p = (f2key(s) >= thr) ? __expf(s - m) : 0.f;
      scrow[c] = p;
      lsum += p;
    }
#pragma unroll
    for (int off = 16; off >= 1; off >>= 1) lsum += __shfl_xor(lsum, off, 32);
    if (lane == 0) rowinv[row] = 1.f / lsum;
  }
  __syncthreads();

  // ---- phase 3: ctx = P @ V (each wave owns 1/4 of the K dimension) ----
  v8f cacc[4] = {};
  for (int ch = wave; ch < T_ / 32; ch += 4) {
    const int k0 = ch * 32;
    const v16bf ap = ld_a_lds_f32(sc + k0, SROW, lane);
#pragma unroll
    for (int dt = 0; dt < 4; ++dt)
      cacc[dt] = wmma_bf16(ap, ld_b16(Vb + (long)dt * 16 * T_ + k0, T_, lane), cacc[dt]);
  }
#pragma unroll
  for (int dt = 0; dt < 4; ++dt)
#pragma unroll
    for (int r = 0; r < 8; ++r)
      red[(wave * 16 + hi * 8 + r) * 64 + dt * 16 + n] = cacc[dt][r];
  __syncthreads();

  // ---- phase 4: reduce wave partials, normalize, emit bf16 ctx in [B,T,E] ----
  {
    const int e0 = threadIdx.x * 8;       // 128 threads x 8 = 16*64 outputs
    const int row = e0 >> 6, d0 = e0 & 63;
    const float inv = rowinv[row];
    v8bf outv;
#pragma unroll
    for (int j = 0; j < 8; ++j) {
      const float s = red[(0 * 16 + row) * 64 + d0 + j] + red[(1 * 16 + row) * 64 + d0 + j]
                    + red[(2 * 16 + row) * 64 + d0 + j] + red[(3 * 16 + row) * 64 + d0 + j];
      outv[j] = (__bf16)(s * inv);
    }
    *(v8bf*)(ctx + ((long)(b * T_ + t0 + row)) * E_ + h * D_ + d0) = outv;
  }
}

// ---------------- host launch ----------------
extern "C" void kernel_launch(void* const* d_in, const int* in_sizes, int n_in,
                              void* d_out, int out_size, void* d_ws, size_t ws_size,
                              hipStream_t stream) {
  (void)in_sizes; (void)n_in; (void)out_size; (void)ws_size;
  const float* x     = (const float*)d_in[0];
  const float* qkv_w = (const float*)d_in[1];
  const float* qkv_b = (const float*)d_in[2];
  const float* out_w = (const float*)d_in[3];
  const float* out_b = (const float*)d_in[4];

  char* ws = (char*)d_ws;
  size_t off = 0;
  auto carve = [&](size_t bytes) -> char* {
    char* p = ws + off;
    off = (off + bytes + 255) & ~(size_t)255;
    return p;
  };
  __bf16* xb   = (__bf16*)carve((size_t)B_ * T_ * E_ * 2);
  __bf16* wqkv = (__bf16*)carve((size_t)E3_ * E_ * 2);
  __bf16* wout = (__bf16*)carve((size_t)E_ * E_ * 2);
  __bf16* qkvb = (__bf16*)carve((size_t)B_ * T_ * E3_ * 2);
  __bf16* vtb  = (__bf16*)carve((size_t)B_ * H_ * D_ * T_ * 2);
  __bf16* ctxb = (__bf16*)carve((size_t)B_ * T_ * E_ * 2);

  cvt_kernel<<<2048, 256, 0, stream>>>(x, xb, (size_t)B_ * T_ * E_);
  cvt_kernel<<<2048, 256, 0, stream>>>(qkv_w, wqkv, (size_t)E3_ * E_);
  cvt_kernel<<<1024, 256, 0, stream>>>(out_w, wout, (size_t)E_ * E_);

  // qkv = x @ qkv_w^T + qkv_b   [4096,1024] x [3072,1024]^T
  gemm_kernel<0><<<dim3(E3_ / 64, (B_ * T_) / 128), 256, 0, stream>>>(
      xb, E_, wqkv, E_, qkv_b, qkvb, E3_, E_);

  vT_kernel<<<(B_ * H_ * D_ * T_) / 256, 256, 0, stream>>>(qkvb, vtb);

  const size_t shmem = (size_t)16 * SROW * 4 + 16384 + 4096 + 16 * 4 + 4 * 4 + 4 * 4;
  attn_kernel<<<dim3(T_ / 16, H_, B_), 128, shmem, stream>>>(qkvb, vtb, ctxb);

  // out = ctx @ out_w^T + out_b   [4096,1024] x [1024,1024]^T -> fp32 d_out
  gemm_kernel<1><<<dim3(E_ / 64, (B_ * T_) / 128), 256, 0, stream>>>(
      ctxb, E_, wout, E_, out_b, d_out, E_, E_);
}
